// MultiHead_66752381714717
// MI455X (gfx1250) — compile-verified, matrix operand order checked by reading, not run on hardware
//
#include <hip/hip_runtime.h>

// ---------------------------------------------------------------------------
// MultiHead attention forward for MI455X (gfx1250, wave32, WMMA bf16 path)
// Outputs (concatenated in d_out): out [B,S,DM] f32, attn [B,H,S,S] f32.
// Scores+softmax+attn@V fully fused through 136KB of LDS per workgroup so
// the 537MB attn tensor crosses HBM exactly once (its mandatory output write).
// All WMMA K-loops are ping-pong double-buffered (no register rotation copies).
// ---------------------------------------------------------------------------

#define Bsz 2
#define Ssz 2048
#define DMsz 1024
#define Hsz 16
#define DKsz 64
#define DVsz 64

typedef __attribute__((ext_vector_type(16))) __bf16 v16bf;
typedef __attribute__((ext_vector_type(8)))  __bf16 v8bf;
typedef __attribute__((ext_vector_type(8)))  float  v8f;
typedef __attribute__((ext_vector_type(4)))  float  v4f;

// Load a 16-element bf16 WMMA fragment from two contiguous 8-element runs.
// CDNA5 16-bit A/B layout: lane<16 holds K=kb..kb+7 and K=16+kb..16+kb+7,
// kb = (lane>=16)?8:0 folded into the pointers by the caller.
static __device__ __forceinline__ v16bf ld_frag(const __bf16* lo, const __bf16* hi) {
    v8bf a0 = *(const v8bf*)lo;
    v8bf a1 = *(const v8bf*)hi;
    v16bf r;
#pragma unroll
    for (int i = 0; i < 8; ++i) { r[i] = a0[i]; r[i + 8] = a1[i]; }
    return r;
}

static __device__ __forceinline__ v8f wmma_bf16(v16bf a, v16bf b, v8f c) {
    return __builtin_amdgcn_wmma_f32_16x16x32_bf16(false, a, false, b, (short)0, c,
                                                   false, false);
}

// ---------------------------------------------------------------------------
__global__ void cvt_f32_bf16(const float* __restrict__ in, __bf16* __restrict__ out, int n) {
    int i = blockIdx.x * blockDim.x + threadIdx.x;
    if (i < n) out[i] = (__bf16)in[i];
}

// Transpose-convert weights: W [K,N] f32 -> WT [N,K] bf16
__global__ void cvt_transpose_w(const float* __restrict__ W, __bf16* __restrict__ WT,
                                int K, int N) {
    int i = blockIdx.x * blockDim.x + threadIdx.x;
    if (i >= K * N) return;
    int n = i / K, k = i % K;
    WT[i] = (__bf16)W[(size_t)k * N + n];
}

// ---------------------------------------------------------------------------
// bf16 GEMM: C[M,N] = A[M,K] * WT[N,K]^T + bias
// Wave tile = 16 rows x 64 cols (4 accumulators, A fragment reused 4x).
// Ping-pong double-buffered K loop (two explicit fragment sets, no copies).
// mode 0: f32 row-major store (final projection)
// mode 1: bf16 scatter to [B,H,S,64]  (q,k : m=b*S+s, n=h*64+d)
// mode 2: bf16 scatter to [B,H,64,S]  (v transposed for attn@V B-fragments)
// ---------------------------------------------------------------------------
__global__ __launch_bounds__(256)
void gemm_bf16(const __bf16* __restrict__ A, const __bf16* __restrict__ WT,
               const float* __restrict__ bias,
               float* __restrict__ outF, __bf16* __restrict__ outB,
               int M, int N, int K, int mode) {
    int wid  = blockIdx.x * (blockDim.x >> 5) + (threadIdx.x >> 5);
    int ngrp = N >> 6;
    int mt = wid / ngrp, ng = wid % ngrp;
    if (mt >= (M >> 4)) return;
    int m0 = mt << 4, n0 = ng << 6;
    int lane = threadIdx.x & 31;
    int lrow = lane & 15;
    int kb   = (lane >> 4) << 3;       // 0 or 8

    const __bf16* arow = A + (size_t)(m0 + lrow) * K;
    const __bf16* wr[4];
#pragma unroll
    for (int c = 0; c < 4; ++c) wr[c] = WT + (size_t)(n0 + c * 16 + lrow) * K;

    v8f acc[4] = {};
    v16bf a0, a1, b0[4], b1[4];

    a0 = ld_frag(arow + kb, arow + 16 + kb);
#pragma unroll
    for (int c = 0; c < 4; ++c) b0[c] = ld_frag(wr[c] + kb, wr[c] + 16 + kb);

    // K assumed a multiple of 64 (K = 1024 here).
    for (int kk = 0; kk < K - 64; kk += 64) {
        a1 = ld_frag(arow + kk + 32 + kb, arow + kk + 48 + kb);
#pragma unroll
        for (int c = 0; c < 4; ++c)
            b1[c] = ld_frag(wr[c] + kk + 32 + kb, wr[c] + kk + 48 + kb);
#pragma unroll
        for (int c = 0; c < 4; ++c) acc[c] = wmma_bf16(a0, b0[c], acc[c]);

        a0 = ld_frag(arow + kk + 64 + kb, arow + kk + 80 + kb);
#pragma unroll
        for (int c = 0; c < 4; ++c)
            b0[c] = ld_frag(wr[c] + kk + 64 + kb, wr[c] + kk + 80 + kb);
#pragma unroll
        for (int c = 0; c < 4; ++c) acc[c] = wmma_bf16(a1, b1[c], acc[c]);
    }
    a1 = ld_frag(arow + K - 32 + kb, arow + K - 16 + kb);
#pragma unroll
    for (int c = 0; c < 4; ++c)
        b1[c] = ld_frag(wr[c] + K - 32 + kb, wr[c] + K - 16 + kb);
#pragma unroll
    for (int c = 0; c < 4; ++c) acc[c] = wmma_bf16(a0, b0[c], acc[c]);
#pragma unroll
    for (int c = 0; c < 4; ++c) acc[c] = wmma_bf16(a1, b1[c], acc[c]);

    int rbase = (lane >> 4) << 3;      // rows 0..7 (lanes 0-15), 8..15 (16-31)
#pragma unroll
    for (int c = 0; c < 4; ++c) {
        int ncol = n0 + c * 16 + lrow;
        float bv = bias ? bias[ncol] : 0.0f;
#pragma unroll
        for (int r = 0; r < 8; ++r) {
            int mrow = m0 + rbase + r;
            float val = acc[c][r] + bv;
            if (mode == 0) {
                outF[(size_t)mrow * N + ncol] = val;
            } else {
                int bb = mrow >> 11, s = mrow & 2047;  // m = b*2048 + s
                int h  = ncol >> 6,  d = ncol & 63;    // n = h*64 + d
                if (mode == 1)
                    outB[(((size_t)(bb * Hsz + h)) * Ssz + s) * DKsz + d] = (__bf16)val;
                else
                    outB[(((size_t)(bb * Hsz + h)) * DVsz + d) * Ssz + s] = (__bf16)val;
            }
        }
    }
}

// ---------------------------------------------------------------------------
// Fully fused attention core. One workgroup (8 waves) per (b,h, 16-row block):
//   phase 1: scores = q k^T / 8, masked, into 128KB LDS strip (WMMA bf16)
//   phase 2: row softmax in LDS; normalized f32 probs -> attn output (HBM)
//   phase 3: ctx = probs @ V straight from LDS (WMMA bf16, 2-way K-split,
//            partials reduced through LDS), ctx stored bf16 [B,S,H*64].
// ---------------------------------------------------------------------------
__global__ __launch_bounds__(256)
void attn_fused(const __bf16* __restrict__ qbf, const __bf16* __restrict__ kbf,
                const __bf16* __restrict__ vT, const unsigned char* __restrict__ mask,
                float* __restrict__ attn, __bf16* __restrict__ ctxb) {
    extern __shared__ float smem[];
    float* sc     = smem;                 // [16][2048] scores/probs
    float* red    = smem + 16 * Ssz;      // [16][16]
    float* rowmax = red + 256;            // [16]
    float* rowsum = rowmax + 16;          // [16]
    float* part   = rowsum + 16;          // [4][256] ctx partial tiles

    int bh = blockIdx.x >> 7;             // 0..31  (b*16+h)
    int qt = blockIdx.x & 127;
    int q0 = qt << 4;
    int bi = bh >> 4;
    int wave = threadIdx.x >> 5;
    int lane = threadIdx.x & 31;
    int lrow = lane & 15;
    int kb   = (lane >> 4) << 3;
    int rbase = (lane >> 4) << 3;

    // ---- phase 1: scores (8 waves x 16 col tiles, ping-pong k-fragments) ----
    const __bf16* qrow = qbf + ((size_t)bh * Ssz + q0 + lrow) * DKsz;
    v16bf a0 = ld_frag(qrow + kb,      qrow + 16 + kb);
    v16bf a1 = ld_frag(qrow + 32 + kb, qrow + 48 + kb);

    const __bf16* kbase = kbf + ((size_t)bh * Ssz + lrow) * DKsz;
    auto ld_k = [&](int j, v16bf& f0, v16bf& f1) {
        const __bf16* kr = kbase + (size_t)((wave + (j << 3)) << 4) * DKsz;
        f0 = ld_frag(kr + kb,      kr + 16 + kb);
        f1 = ld_frag(kr + 32 + kb, kr + 48 + kb);
    };
    auto score_tile = [&](int j, v16bf k0, v16bf k1) {
        v8f acc = {};
        acc = wmma_bf16(a0, k0, acc);
        acc = wmma_bf16(a1, k1, acc);
        int col = ((wave + (j << 3)) << 4) + lrow;
#pragma unroll
        for (int r = 0; r < 8; ++r) {
            int qs = q0 + rbase + r;
            bool mk = mask[((size_t)bi * Ssz + qs) * Ssz + col] != 0;
            sc[(rbase + r) * Ssz + col] = mk ? -1e9f : acc[r] * 0.125f;
        }
    };

    v16bf c0, c1, d0, d1;
    ld_k(0, c0, c1);
#pragma unroll 1
    for (int j = 0; j < 14; j += 2) {
        ld_k(j + 1, d0, d1);
        score_tile(j, c0, c1);
        ld_k(j + 2, c0, c1);
        score_tile(j + 1, d0, d1);
    }
    ld_k(15, d0, d1);
    score_tile(14, c0, c1);
    score_tile(15, d0, d1);
    __syncthreads();

    // ---- phase 2: softmax ----
    int row = threadIdx.x >> 4;           // 0..15
    int sub = threadIdx.x & 15;
    float m = -3.4e38f;
    for (int c = sub; c < Ssz; c += 16) m = fmaxf(m, sc[row * Ssz + c]);
    red[row * 16 + sub] = m;
    __syncthreads();
    if (sub == 0) {
        float mm = red[row * 16];
        for (int j = 1; j < 16; ++j) mm = fmaxf(mm, red[row * 16 + j]);
        rowmax[row] = mm;
    }
    __syncthreads();
    float rm = rowmax[row];
    float s = 0.0f;
    for (int c = sub; c < Ssz; c += 16) {
        float e = __expf(sc[row * Ssz + c] - rm);
        sc[row * Ssz + c] = e;
        s += e;
    }
    red[row * 16 + sub] = s;
    __syncthreads();
    if (sub == 0) {
        float ss = red[row * 16];
        for (int j = 1; j < 16; ++j) ss += red[row * 16 + j];
        rowsum[row] = ss;
    }
    __syncthreads();
    float inv = 1.0f / rowsum[row];
    float* orow = attn + ((size_t)bh * Ssz + q0 + row) * Ssz;
    for (int c = sub; c < Ssz; c += 16) {
        float p = sc[row * Ssz + c] * inv;
        sc[row * Ssz + c] = p;            // normalized probs stay in LDS
        orow[c] = p;                      // mandatory attn output (only HBM pass)
    }
    __syncthreads();

    // ---- phase 3: ctx = probs @ V from LDS (ping-pong V fragments) ----
    int t = wave & 3;                     // n-tile (DV 16-col group)
    int khalf = wave >> 2;                // 2-way K split
    int kstart = khalf << 10;             // 0 or 1024
    const __bf16* vrow = vT + ((size_t)bh * DVsz + (t << 4) + lrow) * Ssz + kstart;
    const float*  al   = sc + lrow * Ssz + kstart;  // this lane's A row in LDS

    auto ld_a = [&](int kk) -> v16bf {
        v4f f0 = *(const v4f*)(al + kk + kb);
        v4f f1 = *(const v4f*)(al + kk + kb + 4);
        v4f f2 = *(const v4f*)(al + kk + 16 + kb);
        v4f f3 = *(const v4f*)(al + kk + 16 + kb + 4);
        v16bf a;
#pragma unroll
        for (int i = 0; i < 4; ++i) {
            a[i]      = (__bf16)f0[i];
            a[4 + i]  = (__bf16)f1[i];
            a[8 + i]  = (__bf16)f2[i];
            a[12 + i] = (__bf16)f3[i];
        }
        return a;
    };

    v8f acc = {};
    v16bf vb0 = ld_frag(vrow + kb, vrow + 16 + kb);
    v16bf vb1;
#pragma unroll 1
    for (int kk = 0; kk < 1024 - 64; kk += 64) {
        vb1 = ld_frag(vrow + kk + 32 + kb, vrow + kk + 48 + kb);
        acc = wmma_bf16(ld_a(kk), vb0, acc);
        vb0 = ld_frag(vrow + kk + 64 + kb, vrow + kk + 80 + kb);
        acc = wmma_bf16(ld_a(kk + 32), vb1, acc);
    }
    vb1 = ld_frag(vrow + 992 + kb, vrow + 1008 + kb);
    acc = wmma_bf16(ld_a(960), vb0, acc);
    acc = wmma_bf16(ld_a(992), vb1, acc);

    if (khalf == 1) {
#pragma unroll
        for (int r = 0; r < 8; ++r) part[(t << 8) + lane * 8 + r] = acc[r];
    }
    __syncthreads();
    if (khalf == 0) {
        int bb = bh >> 4, h = bh & 15;
#pragma unroll
        for (int r = 0; r < 8; ++r) {
            float v = acc[r] + part[(t << 8) + lane * 8 + r];
            int sidx = q0 + rbase + r;
            int dvo  = (t << 4) + lrow;
            ctxb[((size_t)(bb * Ssz + sidx)) * (Hsz * DVsz) + h * DVsz + dvo] = (__bf16)v;
        }
    }
}

// ---------------------------------------------------------------------------
extern "C" void kernel_launch(void* const* d_in, const int* in_sizes, int n_in,
                              void* d_out, int out_size, void* d_ws, size_t ws_size,
                              hipStream_t stream) {
    const float* Q  = (const float*)d_in[0];
    const float* K  = (const float*)d_in[1];
    const float* V  = (const float*)d_in[2];
    const unsigned char* mask = (const unsigned char*)d_in[3];  // jax bool = 1 byte
    const float* wq = (const float*)d_in[4];
    const float* bq = (const float*)d_in[5];
    const float* wk = (const float*)d_in[6];
    const float* bk = (const float*)d_in[7];
    const float* wv = (const float*)d_in[8];
    const float* bv = (const float*)d_in[9];
    const float* wo = (const float*)d_in[10];
    const float* bo = (const float*)d_in[11];

    const size_t E  = (size_t)Bsz * Ssz * DMsz;   // 4,194,304 activation elems
    const size_t WE = (size_t)DMsz * DMsz;        // 1,048,576 weight elems

    __bf16* ws   = (__bf16*)d_ws;                 // total 64 MB of scratch
    __bf16* qbf  = ws;                            // [B,H,S,64]
    __bf16* kbf  = ws + E;                        // [B,H,S,64]
    __bf16* vTbf = ws + 2 * E;                    // [B,H,64,S]
    __bf16* ctxb = ws + 3 * E;                    // [B,S,H*64]
    __bf16* Xq   = ws + 4 * E;                    // Q bf16 [4096,1024]
    __bf16* Xk   = ws + 5 * E;
    __bf16* Xv   = ws + 6 * E;
    __bf16* wqT  = ws + 7 * E;                    // [N,K] transposed weights
    __bf16* wkT  = wqT + WE;
    __bf16* wvT  = wkT + WE;
    __bf16* woT  = wvT + WE;

    float* outMain = (float*)d_out;               // [B,S,DM]
    float* attn    = outMain + E;                 // [B,H,S,S]

    // 1) converts
    int cb = (int)((E + 255) / 256);
    cvt_f32_bf16<<<cb, 256, 0, stream>>>(Q, Xq, (int)E);
    cvt_f32_bf16<<<cb, 256, 0, stream>>>(K, Xk, (int)E);
    cvt_f32_bf16<<<cb, 256, 0, stream>>>(V, Xv, (int)E);
    int wb = (int)((WE + 255) / 256);
    cvt_transpose_w<<<wb, 256, 0, stream>>>(wq, wqT, DMsz, DMsz);
    cvt_transpose_w<<<wb, 256, 0, stream>>>(wk, wkT, DMsz, DMsz);
    cvt_transpose_w<<<wb, 256, 0, stream>>>(wv, wvT, DMsz, DMsz);
    cvt_transpose_w<<<wb, 256, 0, stream>>>(wo, woT, DMsz, DMsz);

    // 2) projections: 4096 tile-waves -> 512 blocks of 8 waves
    gemm_bf16<<<512, 256, 0, stream>>>(Xq, wqT, bq, nullptr, qbf, 4096, 1024, 1024, 1);
    gemm_bf16<<<512, 256, 0, stream>>>(Xk, wkT, bk, nullptr, kbf, 4096, 1024, 1024, 1);
    gemm_bf16<<<512, 256, 0, stream>>>(Xv, wvT, bv, nullptr, vTbf, 4096, 1024, 1024, 2);

    // 3) fused scores + softmax + attn@V : 32 (b,h) x 128 q-blocks
    //    LDS: 16*2048 scores + 256 red + 32 row stats + 1024 ctx partials (f32)
    size_t lds = (size_t)(16 * Ssz + 256 + 32 + 1024) * sizeof(float);
    attn_fused<<<Bsz * Hsz * (Ssz / 16), 256, lds, stream>>>(qbf, kbf, vTbf, mask,
                                                             attn, ctxb);

    // 4) output projection -> f32 out
    gemm_bf16<<<512, 256, 0, stream>>>(ctxb, woT, bo, outMain, nullptr, 4096, 1024, 1024, 0);
}
